// SLNE_Factorized_MultiRate_82643760709710
// MI455X (gfx1250) — matrix operand order repaired and checked by reading, not compiled, not run on hardware
//
#include <hip/hip_runtime.h>
#include <hip/hip_bf16.h>

typedef __attribute__((ext_vector_type(16))) _Float16 v16h;
typedef __attribute__((ext_vector_type(8)))  _Float16 v8h;
typedef __attribute__((ext_vector_type(8)))  float    v8f;

// ---------------------------------------------------------------------------
// Generic gather-GEMM sparse conv:  out = [Σ_k gather(F[,+Fadd], nbr[k]) @ W[k]] + b
// Compile-time CIN/COUT (the network only uses (1,64),(64,64),(64,32),(32,32),(64,1)).
// nbr == nullptr  -> identity gather (dense linear)
// addsrc          -> residual added at the output location
// out_stride/col_off -> strided & channel-offset writes (concat halves, upsample reshape)
// f16 WMMA 16x16x32, f32 accumulate. Two 16-row M tiles per wave, 8 waves/block.
// B fragments staged in LDS in per-lane fragment order (b128 ds traffic only).
// ---------------------------------------------------------------------------
template <int CIN, int COUT>
__global__ __launch_bounds__(256)
void sconv_wmma_kernel(const float* __restrict__ F,
                       const float* __restrict__ Fadd,
                       const int*   __restrict__ nbr,
                       int K, int n_out,
                       const float* __restrict__ W,
                       const float* __restrict__ bias,
                       const float* __restrict__ addsrc,
                       float* __restrict__ out,
                       int out_stride, int col_off, int relu)
{
  constexpr int NT  = (COUT + 15) / 16;   // N tiles (1..4)
  constexpr int NKC = (CIN + 31) / 32;    // K chunks (1..2)

  // B fragments: fragment-ordered, 16 f16 contiguous per (nt, lane)
  __shared__ _Float16 ldsB[NT * 32 * 16];

  const int tid  = threadIdx.x;
  const int wave = tid >> 5;
  const int lane = tid & 31;
  const int m    = lane & 15;
  const int half = lane >> 4;
  const int row0 = (blockIdx.x * 8 + wave) * 32;   // 2 M tiles of 16 rows per wave

  v8f acc[2][NT] = {};

  for (int k = 0; k < K; ++k) {
    // per-k gather indices for both M tiles (idx < 0 => zero row)
    int idxs[2];
    #pragma unroll
    for (int mt = 0; mt < 2; ++mt) {
      int row = row0 + mt * 16 + m;
      idxs[mt] = (row < n_out) ? (nbr ? nbr[(size_t)k * n_out + row] : row) : -1;
    }

    #pragma unroll
    for (int kc = 0; kc < NKC; ++kc) {
      // ---- cooperative stage of B fragments into LDS (f32 -> f16) ----
      __syncthreads();
      {
        int d  = tid >> 1;            // destination (nt*32 + lane), 0..127
        int e0 = (tid & 1) * 8;       // element half (0 or 8)
        if (d < NT * 32) {
          int dl  = d & 31;
          int nt  = d >> 5;
          int col = nt * 16 + (dl & 15);
          int kkb = (dl >> 4) * 8 + e0 * 2;
          v8h wv = {};
          #pragma unroll
          for (int e = 0; e < 8; ++e) {
            int kg = kc * 32 + kkb + e;
            float w = 0.0f;
            if (kg < CIN && col < COUT)    // compile-time folded for full tiles
              w = W[((size_t)k * CIN + kg) * COUT + col];
            wv[e] = (_Float16)w;
          }
          *(v8h*)(&ldsB[d * 16 + e0]) = wv;   // ds_store_b128
        }
      }
      __syncthreads();

      // ---- per-lane A fragments (16x32 f16), vectorized gather ----
      v16h afrag[2];
      #pragma unroll
      for (int mt = 0; mt < 2; ++mt) {
        v16h a = {};
        int idx = idxs[mt];
        if (idx >= 0) {
          const float* src = F + (size_t)idx * CIN + kc * 32;
          if (CIN == 1) {
            // only K=0 is a real input; it lives at half==0, element 0
            if (half == 0) a[0] = (_Float16)src[0];
          } else {
            // two contiguous 8-float runs: K = half*8 + e and K = 16 + half*8 + e
            const float4* r0 = (const float4*)(src + half * 8);
            const float4* r1 = (const float4*)(src + 16 + half * 8);
            float4 p0 = r0[0], p1 = r0[1], p2 = r1[0], p3 = r1[1];
            if (Fadd) {
              const float* s2 = Fadd + (size_t)idx * CIN + kc * 32;
              const float4* q0 = (const float4*)(s2 + half * 8);
              const float4* q1 = (const float4*)(s2 + 16 + half * 8);
              float4 a0 = q0[0], a1 = q0[1], a2 = q1[0], a3 = q1[1];
              p0.x += a0.x; p0.y += a0.y; p0.z += a0.z; p0.w += a0.w;
              p1.x += a1.x; p1.y += a1.y; p1.z += a1.z; p1.w += a1.w;
              p2.x += a2.x; p2.y += a2.y; p2.z += a2.z; p2.w += a2.w;
              p3.x += a3.x; p3.y += a3.y; p3.z += a3.z; p3.w += a3.w;
            }
            a[0]  = (_Float16)p0.x; a[1]  = (_Float16)p0.y;
            a[2]  = (_Float16)p0.z; a[3]  = (_Float16)p0.w;
            a[4]  = (_Float16)p1.x; a[5]  = (_Float16)p1.y;
            a[6]  = (_Float16)p1.z; a[7]  = (_Float16)p1.w;
            a[8]  = (_Float16)p2.x; a[9]  = (_Float16)p2.y;
            a[10] = (_Float16)p2.z; a[11] = (_Float16)p2.w;
            a[12] = (_Float16)p3.x; a[13] = (_Float16)p3.y;
            a[14] = (_Float16)p3.z; a[15] = (_Float16)p3.w;
          }
        }
        afrag[mt] = a;
      }

      // ---- WMMA: each B fragment (2x ds_load_b128) feeds both M tiles ----
      #pragma unroll
      for (int nt = 0; nt < NT; ++nt) {
        v16h b = *(const v16h*)(&ldsB[(nt * 32 + lane) * 16]);
        #pragma unroll
        for (int mt = 0; mt < 2; ++mt) {
          acc[mt][nt] = __builtin_amdgcn_wmma_f32_16x16x32_f16(
              false, afrag[mt], false, b, (short)0, acc[mt][nt], false, false);
        }
      }
    }
  }

  // ---- fused epilogue: bias + optional residual + optional relu ----
  #pragma unroll
  for (int mt = 0; mt < 2; ++mt) {
    #pragma unroll
    for (int nt = 0; nt < NT; ++nt) {
      int col = nt * 16 + m;
      if (COUT < 16 && col >= COUT) continue;   // only the COUT==1 (cls) case
      #pragma unroll
      for (int r = 0; r < 8; ++r) {
        int orow = row0 + mt * 16 + half * 8 + r;
        if (orow < n_out) {
          float v = acc[mt][nt][r] + bias[col];
          size_t o = (size_t)orow * out_stride + col_off + col;
          if (addsrc) v += addsrc[o];
          if (relu) v = fmaxf(v, 0.0f);
          out[o] = v;
        }
      }
    }
  }
}

// ---------------------------------------------------------------------------
// Rate modulation: bmv = exp(relu(th*w1 + b1) @ w2 + b2), C = 64
// ---------------------------------------------------------------------------
__global__ void bm_kernel(const float* __restrict__ th,
                          const float* __restrict__ w1, const float* __restrict__ b1,
                          const float* __restrict__ w2, const float* __restrict__ b2,
                          float* __restrict__ bmv)
{
  __shared__ float h[64];
  int c = threadIdx.x;
  h[c] = fmaxf(th[0] * w1[c] + b1[c], 0.0f);
  __syncthreads();
  float s = b2[c];
  #pragma unroll 8
  for (int j = 0; j < 64; ++j) s += h[j] * w2[j * 64 + c];
  bmv[c] = expf(s);
}

// y_hat = F2*bmv + noise ; F2h = y_hat / bmv
__global__ void scale_noise_kernel(const float* __restrict__ F2,
                                   const float* __restrict__ noise,
                                   const float* __restrict__ bmv,
                                   float* __restrict__ yhat,
                                   float* __restrict__ f2h, int total)
{
  int i = blockIdx.x * blockDim.x + threadIdx.x;
  if (i >= total) return;
  int c = i & 63;
  float s  = bmv[c];
  float yh = F2[i] * s + noise[i];
  yhat[i] = yh;
  f2h[i]  = yh / s;
}

__device__ __forceinline__ float softplusf(float x) { return (x > 20.f) ? x : log1pf(expf(x)); }
__device__ __forceinline__ float sigmoidf(float x)  { return 1.0f / (1.0f + expf(-x)); }

// ---------------------------------------------------------------------------
// EntropyBottleneck likelihood, dims 1->3->3->3->3->1, per (n, c) element.
// A4 exists in the parameter pytree but is unused by the math (mixing is k<n-1).
// ---------------------------------------------------------------------------
__global__ void eb_kernel(const float* __restrict__ yhat,
                          const float* __restrict__ H0, const float* __restrict__ H1,
                          const float* __restrict__ H2, const float* __restrict__ H3,
                          const float* __restrict__ H4,
                          const float* __restrict__ A0, const float* __restrict__ A1,
                          const float* __restrict__ A2, const float* __restrict__ A3,
                          const float* __restrict__ A4,
                          const float* __restrict__ B0, const float* __restrict__ B1,
                          const float* __restrict__ B2, const float* __restrict__ B3,
                          const float* __restrict__ B4,
                          float* __restrict__ lik, int total)
{
  (void)A4;
  int i = blockIdx.x * blockDim.x + threadIdx.x;
  if (i >= total) return;
  int c = i & 63;
  float y = yhat[i];
  float lo[3], hi[3], lo2[3], hi2[3];
  #pragma unroll
  for (int r = 0; r < 3; ++r) {          // layer 0: 1 -> 3, then tanh mixing
    float hw = softplusf(H0[c * 3 + r]);
    float bb = B0[c * 3 + r];
    float ta = tanhf(A0[c * 3 + r]);
    float l  = hw * (y - 0.5f) + bb;
    float u  = hw * (y + 0.5f) + bb;
    lo[r] = l + ta * tanhf(l);
    hi[r] = u + ta * tanhf(u);
  }
  const float* Hs[3] = {H1, H2, H3};
  const float* As[3] = {A1, A2, A3};
  const float* Bs[3] = {B1, B2, B3};
  #pragma unroll
  for (int kl = 0; kl < 3; ++kl) {       // layers 1..3: 3 -> 3, tanh mixing
    #pragma unroll
    for (int r = 0; r < 3; ++r) {
      float l = Bs[kl][c * 3 + r], u = l;
      #pragma unroll
      for (int j = 0; j < 3; ++j) {
        float hw = softplusf(Hs[kl][c * 9 + r * 3 + j]);
        l += hw * lo[j];
        u += hw * hi[j];
      }
      float ta = tanhf(As[kl][c * 3 + r]);
      lo2[r] = l + ta * tanhf(l);
      hi2[r] = u + ta * tanhf(u);
    }
    #pragma unroll
    for (int r = 0; r < 3; ++r) { lo[r] = lo2[r]; hi[r] = hi2[r]; }
  }
  float l = B4[c], u = B4[c];            // layer 4: 3 -> 1
  #pragma unroll
  for (int j = 0; j < 3; ++j) {
    float hw = softplusf(H4[c * 3 + j]);
    l += hw * lo[j];
    u += hw * hi[j];
  }
  float su = l + u;
  float s  = (su > 0.f) ? -1.f : ((su < 0.f) ? 1.f : 0.f);
  float v  = fabsf(sigmoidf(s * u) - sigmoidf(s * l));
  lik[i] = fmaxf(v, 1e-9f);
}

// dst[r, :] = src[idx[r], :]
__global__ void gather_rows_kernel(const float* __restrict__ src,
                                   const int* __restrict__ idx,
                                   float* __restrict__ dst, int nrows, int C)
{
  int i = blockIdx.x * blockDim.x + threadIdx.x;
  if (i >= nrows * C) return;
  int r = i / C, c = i - r * C;
  dst[i] = src[(size_t)idx[r] * C + c];
}

// ---------------------------------------------------------------------------
// Host orchestration. Param flattening (dict keys sorted, lists in order):
//  0 x_F  1 th
//  2..5   bm.{b1,b2,w1,w2}
//  6,7    cls.{W,b}      8,9 dec_conv.{W,b}
//  10..43 dfa0: blocks[i] at 10+10i = {c00.W,c00.b,c01.W,c01.b,c10.W,c10.b,
//         c11.W,c11.b,c12.W,c12.b}; conv1.{W,b}=+30,+31; conv2.{W,b}=+32,+33
//  44..77 dfa1   78..111 dfa2   112..145 dfa3
//  146,147 down1.{W,b}   148,149 down2.{W,b}
//  150..154 eb.H[0..4]  155..159 eb.a[0..4]  160..164 eb.b[0..4]
//  165,166 up.{W,b}
//  167 noise 168 nbr0 169 nbr_d1 170 nbr1 171 nbr_d2 172 nbr2 173 nbr_up
//  174 sort_perm 175 keep_idx
// ---------------------------------------------------------------------------
extern "C" void kernel_launch(void* const* d_in, const int* in_sizes, int n_in,
                              void* d_out, int out_size, void* d_ws, size_t ws_size,
                              hipStream_t stream)
{
  (void)n_in; (void)out_size; (void)ws_size;
  auto P = [&](int i) { return (const float*)d_in[i]; };
  auto I = [&](int i) { return (const int*)d_in[i]; };

  const int n0     = in_sizes[0];
  const int n1     = in_sizes[169] / 8;
  const int n2     = in_sizes[167] / 64;
  const int n_up   = in_sizes[174];
  const int n_keep = in_sizes[175];

  const int* nbr0   = I(168);
  const int* nbr_d1 = I(169);
  const int* nbr1   = I(170);
  const int* nbr_d2 = I(171);
  const int* nbr2   = I(172);
  const int* nbr_up = I(173);
  const int* sperm  = I(174);
  const int* keep   = I(175);

  int nmax = n0;
  if (n1 > nmax) nmax = n1;
  if (n_up > nmax) nmax = n_up;

  float* ws = (float*)d_ws;
  size_t off = 0;
  float* X    = ws + off; off += (size_t)nmax * 64;
  float* Y    = ws + off; off += (size_t)nmax * 64;
  float* Z    = ws + off; off += (size_t)nmax * 64;
  float* T0   = ws + off; off += (size_t)nmax * 32;
  float* T1   = ws + off; off += (size_t)nmax * 32;
  float* bmv  = ws + off; off += 64;
  float* yhat = ws + off; off += (size_t)n2 * 64;
  float* f2h  = ws + off; off += (size_t)n2 * 64;

  float* fout_out = (float*)d_out;                     // (n_keep, 64)
  float* lik_out  = fout_out + (size_t)n_keep * 64;    // (n2, 64)
  float* cls_out  = lik_out + (size_t)n2 * 64;         // (n_up, 1)

  auto sconv = [&](const float* Fin, const float* Fadd, const int* nbr, int K,
                   int n_o, int cin, int cout, const float* Wp, const float* bp,
                   const float* addsrc, float* out, int ostride, int coff, int relu) {
    dim3 grid((n_o + 255) / 256);
    if (cin == 64 && cout == 64)
      sconv_wmma_kernel<64, 64><<<grid, 256, 0, stream>>>(Fin, Fadd, nbr, K, n_o,
          Wp, bp, addsrc, out, ostride, coff, relu);
    else if (cin == 64 && cout == 32)
      sconv_wmma_kernel<64, 32><<<grid, 256, 0, stream>>>(Fin, Fadd, nbr, K, n_o,
          Wp, bp, addsrc, out, ostride, coff, relu);
    else if (cin == 32 && cout == 32)
      sconv_wmma_kernel<32, 32><<<grid, 256, 0, stream>>>(Fin, Fadd, nbr, K, n_o,
          Wp, bp, addsrc, out, ostride, coff, relu);
    else if (cin == 1 && cout == 64)
      sconv_wmma_kernel<1, 64><<<grid, 256, 0, stream>>>(Fin, Fadd, nbr, K, n_o,
          Wp, bp, addsrc, out, ostride, coff, relu);
    else if (cin == 64 && cout == 1)
      sconv_wmma_kernel<64, 1><<<grid, 256, 0, stream>>>(Fin, Fadd, nbr, K, n_o,
          Wp, bp, addsrc, out, ostride, coff, relu);
  };

  // IRN: out[:,0:32] = sconv(relu(sconv(F,c00)),c01)+F[:,0:32]
  //      out[:,32:64] = lin(relu(sconv(relu(lin(F,c10)),c11)),c12)+F[:,32:64]
  auto irn = [&](int R, const float* Fin, float* out, const int* nbr, int n_o) {
    sconv(Fin, nullptr, nbr,     27, n_o, 64, 32, P(R + 0), P(R + 1), nullptr, T0, 32, 0, 1);
    sconv(T0,  nullptr, nbr,     27, n_o, 32, 32, P(R + 2), P(R + 3), Fin,     out, 64, 0, 0);
    sconv(Fin, nullptr, nullptr,  1, n_o, 64, 32, P(R + 4), P(R + 5), nullptr, T1, 32, 0, 1);
    sconv(T1,  nullptr, nbr,     27, n_o, 32, 32, P(R + 6), P(R + 7), nullptr, T0, 32, 0, 1);
    sconv(T0,  nullptr, nullptr,  1, n_o, 32, 32, P(R + 8), P(R + 9), Fin,     out, 64, 32, 0);
  };

  // DFA: x=relu(conv1(F)); h=irn^3(x); out=conv2(h+x). Result lands in Rb.
  auto dfa = [&](int B, const float* Fin, int cin, const int* nbr, int n_o,
                 float* Pb, float* Qb, float* Rb) {
    sconv(Fin, nullptr, nbr, 27, n_o, cin, 64, P(B + 30), P(B + 31), nullptr, Pb, 64, 0, 1);
    irn(B + 0,  Pb, Qb, nbr, n_o);
    irn(B + 10, Qb, Rb, nbr, n_o);
    irn(B + 20, Rb, Qb, nbr, n_o);
    sconv(Qb, Pb, nbr, 27, n_o, 64, 64, P(B + 32), P(B + 33), nullptr, Rb, 64, 0, 0);
  };

  // ----- encoder -----
  dfa(10, P(0), 1, nbr0, n0, X, Y, Z);                                            // F0 -> Z
  sconv(Z, nullptr, nbr_d1, 8, n1, 64, 64, P(146), P(147), nullptr, X, 64, 0, 0); // down1 -> X
  dfa(44, X, 64, nbr1, n1, Y, Z, X);                                              // F1 -> X
  sconv(X, nullptr, nbr_d2, 8, n2, 64, 64, P(148), P(149), nullptr, Y, 64, 0, 0); // down2 -> Y
  dfa(78, Y, 64, nbr2, n2, Z, X, Y);                                              // F2 -> Y

  // ----- bottleneck -----
  bm_kernel<<<1, 64, 0, stream>>>(P(1), P(4), P(2), P(5), P(3), bmv);
  int tot2 = n2 * 64;
  scale_noise_kernel<<<(tot2 + 255) / 256, 256, 0, stream>>>(Y, P(167), bmv, yhat, f2h, tot2);
  eb_kernel<<<(tot2 + 255) / 256, 256, 0, stream>>>(yhat,
      P(150), P(151), P(152), P(153), P(154),
      P(155), P(156), P(157), P(158), P(159),
      P(160), P(161), P(162), P(163), P(164),
      lik_out, tot2);

  // ----- decoder -----
  // generative upsample: pre[n*8+kk, :] = F2h[n] @ Wu[kk] + b  (strided writes into X)
  for (int kk = 0; kk < 8; ++kk)
    sconv(f2h, nullptr, nullptr, 1, n2, 64, 64, P(165) + (size_t)kk * 64 * 64, P(166),
          nullptr, X, 512, kk * 64, 0);
  int totu = n_up * 64;
  gather_rows_kernel<<<(totu + 255) / 256, 256, 0, stream>>>(X, sperm, Y, n_up, 64); // sort

  dfa(112, Y, 64, nbr_up, n_up, X, Z, Y);                                           // dfa3 -> Y
  sconv(Y, nullptr, nbr_up, 27, n_up, 64, 64, P(8), P(9), nullptr, Z, 64, 0, 0);    // dec_conv -> Z
  sconv(Z, nullptr, nbr_up, 27, n_up, 64, 1,  P(6), P(7), nullptr, cls_out, 1, 0, 0); // cls
  int totk = n_keep * 64;
  gather_rows_kernel<<<(totk + 255) / 256, 256, 0, stream>>>(Z, keep, fout_out, n_keep, 64);
}
